// GATModel_Softmax_4535485465120
// MI455X (gfx1250) — matrix operand order repaired and checked by reading, not compiled, not run on hardware
//
#include <hip/hip_runtime.h>
#include <hip/hip_bf16.h>

// ---------------- problem constants ----------------
#define NN    8000
#define MPAD  8064          // 63 * 128 row padding for GEMM A operands
#define DIN   1024
#define HH    3
#define CC    1024
#define HC    (HH*CC)       // 3072
#define NCLS  460
#define NCLSP 512           // 4 * 128 col padding for FC GEMM
#define EE    40000
#define ETOT  (EE+NN)       // 48000 edges incl. self loops
#define NEG_SLOPE 0.2f

typedef __attribute__((ext_vector_type(16))) __bf16 v16bf;
typedef __attribute__((ext_vector_type(8)))  __bf16 v8bf;
typedef __attribute__((ext_vector_type(8)))  float  v8f;
typedef unsigned short u16;
typedef unsigned int   u32;

// ---------------- helpers ----------------
__device__ __forceinline__ u16 f2bf(float f) {
    u32 u = __float_as_uint(f);
    u32 r = u + 0x7FFFu + ((u >> 16) & 1u);   // round-nearest-even
    return (u16)(r >> 16);
}
// monotone float<->uint encoding for atomicMax on signed floats
__device__ __forceinline__ u32 enc_f(float f) {
    u32 b = __float_as_uint(f);
    return b ^ ((b >> 31) ? 0xFFFFFFFFu : 0x80000000u);
}
__device__ __forceinline__ float dec_f(u32 e) {
    u32 b = (e & 0x80000000u) ? (e ^ 0x80000000u) : ~e;
    return __uint_as_float(b);
}

// ---------------- trivial elementwise kernels ----------------
__global__ void k_zero_f32(float* p, int n) {
    for (int i = blockIdx.x * blockDim.x + threadIdx.x; i < n; i += gridDim.x * blockDim.x)
        p[i] = 0.0f;
}
__global__ void k_cvt_bf16(const float* __restrict__ in, u16* __restrict__ out, int n) {
    for (int i = blockIdx.x * blockDim.x + threadIdx.x; i < n; i += gridDim.x * blockDim.x)
        out[i] = f2bf(in[i]);
}
// transpose + convert:  out[n*K + k] = bf16(in[k*N + n]);  i indexes n*K+k (coalesced writes)
__global__ void k_cvt_bf16_t(const float* __restrict__ in, u16* __restrict__ out,
                             int K, int N) {
    const int total = N * K;
    for (int i = blockIdx.x * blockDim.x + threadIdx.x; i < total; i += gridDim.x * blockDim.x) {
        const int k = i & (K - 1);       // K is a power of two (1024)
        const int n = i / K;
        out[i] = f2bf(in[(size_t)k * N + n]);
    }
}
// out_mean + bias -> bf16  (bias indexed by col % CC)
__global__ void k_biascvt(const float* __restrict__ in, const float* __restrict__ bias,
                          u16* __restrict__ out, int n) {
    for (int i = blockIdx.x * blockDim.x + threadIdx.x; i < n; i += gridDim.x * blockDim.x)
        out[i] = f2bf(in[i] + bias[i & (CC - 1)]);
}

// ---------------- bf16 WMMA GEMM ----------------
//   C[M x Nreal] (leading dim Cld) = A[Mpad x K] * Bt[Npad x K]^T + bias[Nreal]
// Register-pipelined + LDS double-buffered: next K-tile is fetched to VGPRs
// while WMMAs run on the current tile; one barrier per K-step. Hot loop per
// thread: 4x b128 global loads (overlapped), 4x b128 LDS stores, 12x b128
// LDS loads, 8x v_wmma_f32_16x16x32_bf16. No bounds checks (padded operands).
#define BM 128
#define BN 128
#define BK 32
#define LDSBUF (BM * BK)    // elements per A (or B) buffer

__global__ __launch_bounds__(256)
void k_gemm_bf16(const u16* __restrict__ A, const u16* __restrict__ Bt,
                 const float* __restrict__ bias, float* __restrict__ C,
                 int M, int Nreal, int Cld, int K)
{
    __shared__ alignas(64) u16 sA [2 * LDSBUF];   // [buf][row][k]
    __shared__ alignas(64) u16 sBt[2 * LDSBUF];   // [buf][col][k]

    const int t    = threadIdx.x;
    const int lane = t & 31;
    const int wid  = t >> 5;
    const int hi   = (lane >> 4) & 1;   // half-wave select
    const int lrow = lane & 15;
    const int wm0  = (wid & 3) * 32;    // wave row offset in block tile
    const int wn0  = (wid >> 2) * 64;   // wave col offset in block tile
    const int m0   = blockIdx.y * BM;
    const int n0   = blockIdx.x * BN;

    // staging coordinates: each thread moves 2 A-rows and 2 B-cols, 8 bf16 each
    const int srow = t >> 2;            // 0..63
    const int k8   = (t & 3) * 8;       // 0,8,16,24
    const u16* gA0 = A  + (size_t)(m0 + srow)      * K + k8;
    const u16* gA1 = A  + (size_t)(m0 + srow + 64) * K + k8;
    const u16* gB0 = Bt + (size_t)(n0 + srow)      * K + k8;
    const u16* gB1 = Bt + (size_t)(n0 + srow + 64) * K + k8;

    const v8f vzero = {0.f,0.f,0.f,0.f,0.f,0.f,0.f,0.f};
    v8f acc[2][4];
    for (int i = 0; i < 2; ++i)
        for (int j = 0; j < 4; ++j)
            acc[i][j] = vzero;

    // preload K-tile 0 into registers
    uint4 ra0 = *(const uint4*)(gA0);
    uint4 ra1 = *(const uint4*)(gA1);
    uint4 rb0 = *(const uint4*)(gB0);
    uint4 rb1 = *(const uint4*)(gB1);

    int p = 0;
    for (int k0 = 0; k0 < K; k0 += BK, p ^= 1) {
        // ---- publish current tile from registers into LDS buffer p ----
        u16* bA = sA  + p * LDSBUF;
        u16* bB = sBt + p * LDSBUF;
        *(uint4*)(&bA[ srow       * BK + k8]) = ra0;
        *(uint4*)(&bA[(srow + 64) * BK + k8]) = ra1;
        *(uint4*)(&bB[ srow       * BK + k8]) = rb0;
        *(uint4*)(&bB[(srow + 64) * BK + k8]) = rb1;
        __syncthreads();

        // ---- issue next tile's global loads (latency hidden by WMMAs) ----
        if (k0 + BK < K) {
            ra0 = *(const uint4*)(gA0 + k0 + BK);
            ra1 = *(const uint4*)(gA1 + k0 + BK);
            rb0 = *(const uint4*)(gB0 + k0 + BK);
            rb1 = *(const uint4*)(gB1 + k0 + BK);
        }

        // ---- fragments per ISA 16-bit A / B lane layouts ----
        v16bf af[2], bfr[4];
        #pragma unroll
        for (int wm = 0; wm < 2; ++wm) {
            const int row = wm0 + wm * 16 + lrow;
            // lanes<16: K 0..7 & 16..23 ; lanes>=16: K 8..15 & 24..31
            const v8bf lo  = *(const v8bf*)&bA[row * BK + hi * 8];
            const v8bf hi8 = *(const v8bf*)&bA[row * BK + 16 + hi * 8];
            af[wm] = __builtin_shufflevector(lo, hi8,
                        0,1,2,3,4,5,6,7,8,9,10,11,12,13,14,15);
        }
        #pragma unroll
        for (int wn = 0; wn < 4; ++wn) {
            const int coln = wn0 + wn * 16 + lrow;
            // lane = N, values = K: lanes<16 K 0..7 & 16..23 ; lanes>=16: K 8..15 & 24..31
            const v8bf blo = *(const v8bf*)&bB[coln * BK + hi * 8];
            const v8bf bhi = *(const v8bf*)&bB[coln * BK + 16 + hi * 8];
            bfr[wn] = __builtin_shufflevector(blo, bhi,
                        0,1,2,3,4,5,6,7,8,9,10,11,12,13,14,15);
        }
        #pragma unroll
        for (int wm = 0; wm < 2; ++wm)
            #pragma unroll
            for (int wn = 0; wn < 4; ++wn)
                acc[wm][wn] = __builtin_amdgcn_wmma_f32_16x16x32_bf16(
                    false, af[wm], false, bfr[wn],
                    (short)0, acc[wm][wn], false, false);
        // no trailing barrier: next store targets the other LDS buffer, and
        // the next iteration's barrier orders buffer reuse two steps apart
    }

    // store: C/D layout lane=N, VGPR v -> M = v + hi*8
    for (int wm = 0; wm < 2; ++wm)
        for (int wn = 0; wn < 4; ++wn) {
            const int gcol = n0 + wn0 + wn * 16 + lrow;
            if (gcol >= Nreal) continue;
            const float bb = bias ? bias[gcol] : 0.f;
            #pragma unroll
            for (int v = 0; v < 8; ++v) {
                const int grow = m0 + wm0 + wm * 16 + hi * 8 + v;
                if (grow < M)
                    C[(size_t)grow * Cld + gcol] = acc[wm][wn][v] + bb;
            }
        }
}

// ---------------- edge scores: e[h] = att[h] . leaky_relu(xl[src]+xr[dst]) ----------------
// one wave per edge (incl. self loops); segment-max into m_enc via atomicMax
__global__ __launch_bounds__(256)
void k_edge_score(const int* __restrict__ ei, const float* __restrict__ xl,
                  const float* __restrict__ xr, const float* __restrict__ att,
                  float* __restrict__ escore, u32* __restrict__ m_enc)
{
    const int lane = threadIdx.x & 31;
    const int wid  = threadIdx.x >> 5;
    const int e    = blockIdx.x * 8 + wid;
    if (e >= ETOT) return;
    const int s = (e < EE) ? ei[2 * e]     : (e - EE);
    const int d = (e < EE) ? ei[2 * e + 1] : (e - EE);

    for (int h = 0; h < HH; ++h) {
        const float* pl = xl + (size_t)s * HC + h * CC;
        const float* pr = xr + (size_t)d * HC + h * CC;
        const float* pa = att + h * CC;
        float acc = 0.f;
        for (int c = lane; c < CC; c += 32) {
            float v = pl[c] + pr[c];
            v = (v > 0.f) ? v : NEG_SLOPE * v;
            acc += pa[c] * v;
        }
        #pragma unroll
        for (int off = 16; off > 0; off >>= 1)
            acc += __shfl_xor(acc, off, 32);
        if (lane == 0) {
            escore[(size_t)e * HH + h] = acc;
            atomicMax(&m_enc[(size_t)d * HH + h], enc_f(acc));
        }
    }
}

// ---------------- exp(e - m[dst]) and segment-sum denominator ----------------
__global__ void k_exp_denom(const int* __restrict__ ei, float* __restrict__ escore,
                            const u32* __restrict__ m_enc, float* __restrict__ denom)
{
    const int i = blockIdx.x * blockDim.x + threadIdx.x;
    if (i >= ETOT * HH) return;
    const int e = i / HH, h = i % HH;
    const int d = (e < EE) ? ei[2 * e + 1] : (e - EE);
    const float m = dec_f(m_enc[(size_t)d * HH + h]);
    const float a = __expf(escore[i] - m);
    escore[i] = a;                         // overwrite score with numerator
    atomicAdd(&denom[(size_t)d * HH + h], a);
}

// ---------------- weighted scatter, fused mean over heads ----------------
// one block per edge: out_mean[dst][c] += sum_h alpha[h]/H * xl[src][h][c]
__global__ __launch_bounds__(256)
void k_scatter(const int* __restrict__ ei, const float* __restrict__ escore,
               const float* __restrict__ denom, const float* __restrict__ xl,
               float* __restrict__ out_mean)
{
    __shared__ float s_alpha[HH];
    const int e = blockIdx.x;
    const int t = threadIdx.x;
    const int s = (e < EE) ? ei[2 * e]     : (e - EE);
    const int d = (e < EE) ? ei[2 * e + 1] : (e - EE);
    if (t < HH) {
        const float a   = escore[(size_t)e * HH + t];
        const float den = denom[(size_t)d * HH + t];
        s_alpha[t] = a / (den + 1e-16f) * (1.0f / HH);
    }
    __syncthreads();
    const float* px = xl + (size_t)s * HC;
    for (int c = t; c < CC; c += 256) {
        const float v = s_alpha[0] * px[c]
                      + s_alpha[1] * px[CC + c]
                      + s_alpha[2] * px[2 * CC + c];
        atomicAdd(&out_mean[(size_t)d * CC + c], v);
    }
}

// ---------------- row softmax over NCLS (logits ld = NCLSP), one wave per row ----------------
__global__ __launch_bounds__(256)
void k_softmax(const float* __restrict__ logits, float* __restrict__ out)
{
    const int lane = threadIdx.x & 31;
    const int wid  = threadIdx.x >> 5;
    const int row  = blockIdx.x * 8 + wid;
    if (row >= NN) return;
    const float* p = logits + (size_t)row * NCLSP;
    float mx = -3.4e38f;
    for (int c = lane; c < NCLS; c += 32) mx = fmaxf(mx, p[c]);
    #pragma unroll
    for (int off = 16; off > 0; off >>= 1) mx = fmaxf(mx, __shfl_xor(mx, off, 32));
    float s = 0.f;
    for (int c = lane; c < NCLS; c += 32) s += __expf(p[c] - mx);
    #pragma unroll
    for (int off = 16; off > 0; off >>= 1) s += __shfl_xor(s, off, 32);
    const float inv = 1.0f / s;
    for (int c = lane; c < NCLS; c += 32)
        out[(size_t)row * NCLS + c] = __expf(p[c] - mx) * inv;
}

// ---------------- host-side orchestration ----------------
extern "C" void kernel_launch(void* const* d_in, const int* in_sizes, int n_in,
                              void* d_out, int out_size, void* d_ws, size_t ws_size,
                              hipStream_t stream)
{
    (void)in_sizes; (void)n_in; (void)out_size; (void)ws_size;
    const float* x    = (const float*)d_in[0];
    const int*   ei   = (const int*)  d_in[1];
    const float* Wl   = (const float*)d_in[2];
    const float* bl   = (const float*)d_in[3];
    const float* Wr   = (const float*)d_in[4];
    const float* br   = (const float*)d_in[5];
    const float* att  = (const float*)d_in[6];
    const float* bias = (const float*)d_in[7];
    const float* Wf   = (const float*)d_in[8];
    const float* bfv  = (const float*)d_in[9];
    float* out = (float*)d_out;

    // carve workspace (256B aligned slabs)
    size_t off = 0;
    auto carve = [&](size_t bytes) -> void* {
        void* p = (char*)d_ws + off;
        off += (bytes + 255) & ~(size_t)255;
        return p;
    };
    u16*   xbf   = (u16*)  carve((size_t)MPAD * DIN * 2);     // row-padded A
    u16*   WlT   = (u16*)  carve((size_t)HC * DIN * 2);       // [3072][1024] bf16 (transposed)
    u16*   WrT   = (u16*)  carve((size_t)HC * DIN * 2);
    u16*   WfT   = (u16*)  carve((size_t)NCLSP * CC * 2);     // [512][1024], zero padded
    float* xl    = (float*)carve((size_t)NN * HC * 4);
    float* xr    = (float*)carve((size_t)NN * HC * 4);
    float* esc   = (float*)carve((size_t)ETOT * HH * 4);
    u32*   menc  = (u32*)  carve((size_t)NN * HH * 4);
    float* denom = (float*)carve((size_t)NN * HH * 4);
    float* omean = (float*)carve((size_t)NN * CC * 4);
    u16*   hbf   = (u16*)  carve((size_t)MPAD * CC * 2);      // row-padded A for FC
    float* logit = (float*)carve((size_t)NN * NCLSP * 4);

    // 1) zero-init padded bf16 buffers + accumulators (menc zero == encoded minimum)
    k_zero_f32<<<4096, 256, 0, stream>>>((float*)xbf, MPAD * DIN / 2);
    k_zero_f32<<<4096, 256, 0, stream>>>((float*)hbf, MPAD * CC / 2);
    k_zero_f32<<<1024, 256, 0, stream>>>((float*)WfT, NCLSP * CC / 2);
    k_zero_f32<<<4096, 256, 0, stream>>>(omean, NN * CC);
    k_zero_f32<<<64,   256, 0, stream>>>(denom, NN * HH);
    k_zero_f32<<<64,   256, 0, stream>>>((float*)menc, NN * HH);

    // 2) converts: x row-major; weights transposed to [N][K] bf16
    k_cvt_bf16  <<<4096, 256, 0, stream>>>(x, xbf, NN * DIN);
    k_cvt_bf16_t<<<4096, 256, 0, stream>>>(Wl, WlT, DIN, HC);
    k_cvt_bf16_t<<<4096, 256, 0, stream>>>(Wr, WrT, DIN, HC);
    k_cvt_bf16_t<<<2048, 256, 0, stream>>>(Wf, WfT, CC, NCLS);

    // 3) projection GEMMs: xl = x@Wl + bl ; xr = x@Wr + br
    dim3 g1(HC / BN, MPAD / BM);
    k_gemm_bf16<<<g1, 256, 0, stream>>>(xbf, WlT, bl, xl, NN, HC, HC, DIN);
    k_gemm_bf16<<<g1, 256, 0, stream>>>(xbf, WrT, br, xr, NN, HC, HC, DIN);

    // 4) edge attention scores + segment max
    k_edge_score<<<(ETOT + 7) / 8, 256, 0, stream>>>(ei, xl, xr, att, esc, menc);

    // 5) numerators + segment sum
    k_exp_denom<<<(ETOT * HH + 255) / 256, 256, 0, stream>>>(ei, esc, menc, denom);

    // 6) weighted scatter + head mean
    k_scatter<<<ETOT, 256, 0, stream>>>(ei, esc, denom, xl, omean);

    // 7) + conv bias, convert to bf16 for FC
    k_biascvt<<<4096, 256, 0, stream>>>(omean, bias, hbf, NN * CC);

    // 8) FC GEMM: logits[8000x460 in ld 512] = h @ Wf + bf
    dim3 g3(NCLSP / BN, MPAD / BM);
    k_gemm_bf16<<<g3, 256, 0, stream>>>(hbf, WfT, bfv, logit, NN, NCLS, NCLSP, CC);

    // 9) row softmax -> d_out
    k_softmax<<<(NN + 7) / 8, 256, 0, stream>>>(logit, out);
}
